// GIN_ZINC_v2_77008763617630
// MI455X (gfx1250) — matrix operand order repaired and checked by reading, not compiled, not run on hardware
//
#include <hip/hip_runtime.h>
#include <hip/hip_bf16.h>

typedef __attribute__((ext_vector_type(16))) _Float16 v16h;
typedef __attribute__((ext_vector_type(8)))  float    v8f;
typedef __attribute__((ext_vector_type(4)))  float    v4f;

union V16HU { v16h h; v4f f[2]; };

#define BN_EPS 1e-5f

// ---------------------------------------------------------------------------
// Generic utility kernels
// ---------------------------------------------------------------------------
__global__ void k_zero(float* p, int n) {
    int i = blockIdx.x * blockDim.x + threadIdx.x;
    if (i < n) p[i] = 0.f;
}

// ---------------------------------------------------------------------------
// PE MLP: hp = relu(pe@W1+b1)@W2+b2, plus per-channel sum/sumsq for BN
// ---------------------------------------------------------------------------
__global__ void k_pe(const float* __restrict__ pe,
                     const float* __restrict__ W1, const float* __restrict__ b1,
                     const float* __restrict__ W2, const float* __restrict__ b2,
                     float* __restrict__ hp, float* __restrict__ sums, int N) {
    __shared__ float s_sum[16], s_sq[16];
    if (threadIdx.x < 16) { s_sum[threadIdx.x] = 0.f; s_sq[threadIdx.x] = 0.f; }
    __syncthreads();
    int n = blockIdx.x * blockDim.x + threadIdx.x;
    if (n < N) {
        float pin[20];
#pragma unroll
        for (int i = 0; i < 20; ++i) pin[i] = pe[n * 20 + i];
        float t[16];
#pragma unroll
        for (int j = 0; j < 16; ++j) {
            float s = b1[j];
#pragma unroll
            for (int i = 0; i < 20; ++i) s += pin[i] * W1[i * 16 + j];
            t[j] = fmaxf(s, 0.f);
        }
#pragma unroll
        for (int j = 0; j < 16; ++j) {
            float s = b2[j];
#pragma unroll
            for (int k = 0; k < 16; ++k) s += t[k] * W2[k * 16 + j];
            hp[n * 16 + j] = s;
            atomicAdd(&s_sum[j], s);
            atomicAdd(&s_sq[j], s * s);
        }
    }
    __syncthreads();
    if (threadIdx.x < 16) {
        atomicAdd(&sums[threadIdx.x],      s_sum[threadIdx.x]);
        atomicAdd(&sums[16 + threadIdx.x], s_sq[threadIdx.x]);
    }
}

__global__ void k_pe_fin(const float* __restrict__ sums,
                         const float* __restrict__ gamma, const float* __restrict__ beta,
                         float* __restrict__ ss, int N) {
    int c = threadIdx.x;  // 16 threads
    float inv = 1.f / (float)N;
    float mean = sums[c] * inv;
    float var  = sums[16 + c] * inv - mean * mean;
    float sc   = gamma[c] * rsqrtf(var + BN_EPS);
    ss[c]      = sc;
    ss[16 + c] = beta[c] - mean * sc;
}

// ---------------------------------------------------------------------------
// Build h_in[N,96] = [atom_emb[x] (64) | bn(hp) (16) | zeros (16)]
// ---------------------------------------------------------------------------
__global__ void k_embed(const int* __restrict__ x, const float* __restrict__ atom_emb,
                        const float* __restrict__ hp, const float* __restrict__ ss,
                        float* __restrict__ h_in) {
    int n = blockIdx.x;
    int j = threadIdx.x;  // 96 threads
    float v;
    if (j < 64)       v = atom_emb[x[n] * 64 + j];
    else if (j < 80)  { int c = j - 64; v = hp[n * 16 + c] * ss[c] + ss[16 + c]; }
    else              v = 0.f;
    h_in[(size_t)n * 96 + j] = v;
}

// ---------------------------------------------------------------------------
// Pack a row-major f32 weight [Ksrc x 128] into WMMA B-operand order (f16).
// Lane l of the wave needs B[koff + (j&7) + (j&8 ? 16 : 0)][col] for j=0..15, where
// col = 16*ct + (l&15), koff = ks*32 + (l<16 ? 0 : 8).
// ---------------------------------------------------------------------------
__global__ void k_pack(const float* __restrict__ W, _Float16* __restrict__ out,
                       int Ksrc, int ksteps) {
    int tid = blockIdx.x * blockDim.x + threadIdx.x;
    int total = 8 * ksteps * 512;
    if (tid >= total) return;
    int j    = tid & 15;
    int lane = (tid >> 4) & 31;
    int ks   = (tid >> 9) % ksteps;
    int ct   = tid / (ksteps << 9);
    int k    = ks * 32 + ((lane & 16) ? 8 : 0) + (j & 7) + ((j & 8) ? 16 : 0);
    int col  = ct * 16 + (lane & 15);
    float v  = (k < Ksrc) ? W[k * 128 + col] : 0.f;
    out[tid] = (_Float16)v;
}

// ---------------------------------------------------------------------------
// WMMA GEMM: C[N,128] = A[N,K] * Bpacked + bias, optional relu.
// 4 waves / block; B (K x 128, packed) staged once into LDS; each wave
// computes a full 16x128 output strip (8 accumulators). K fully unrolled.
// AHALF=1: A is f16 row-major. AHALF=0: A is f32 row-major (converted inline).
// ---------------------------------------------------------------------------
template <int AHALF, int KSTEPS>
__global__ __launch_bounds__(128) void gemm_wmma(
        const void* __restrict__ A_, const _Float16* __restrict__ Bp,
        const float* __restrict__ bias,
        float* __restrict__ outF, _Float16* __restrict__ outH,
        int nRowTiles, int relu) {
    __shared__ __align__(16) _Float16 Bs[8 * KSTEPS * 512];

    // Cooperative stage of packed B into LDS (layout identical to global pack)
    {
        const int4* src = (const int4*)Bp;
        int4*       dst = (int4*)Bs;
        const int nvec = (8 * KSTEPS * 512) / 8;  // 8 halves per int4
        for (int i = threadIdx.x; i < nvec; i += 128) dst[i] = src[i];
    }
    __syncthreads();

    const int lane = threadIdx.x & 31;
    const int wv   = threadIdx.x >> 5;             // wave in block: 0..3
    const int rt   = blockIdx.x * 4 + wv;          // row tile
    if (rt >= nRowTiles) return;                   // wave-uniform exit (after barrier)

    const int ln15 = lane & 15;
    const int hi16 = (lane >> 4) & 1;
    const int row  = rt * 16 + ln15;               // A row this lane supplies
    const int K    = KSTEPS * 32;

    v8f acc[8];
#pragma unroll
    for (int ct = 0; ct < 8; ++ct) {
        float bv = bias[ct * 16 + ln15];
#pragma unroll
        for (int v = 0; v < 8; ++v) acc[ct][v] = bv;
    }

#pragma unroll
    for (int ks = 0; ks < KSTEPS; ++ks) {
        const int kb = ks * 32 + hi16 * 8;
        v16h a;
        if (AHALF) {
            const _Float16* A = (const _Float16*)A_;
            V16HU u;
            u.f[0] = *(const v4f*)(A + (size_t)row * K + kb);        // k = kb..kb+7
            u.f[1] = *(const v4f*)(A + (size_t)row * K + kb + 16);   // k = kb+16..kb+23
            a = u.h;
        } else {
            const float* A = (const float*)A_;
            const float* p = A + (size_t)row * K + kb;
            v4f f0 = *(const v4f*)(p);
            v4f f1 = *(const v4f*)(p + 4);
            v4f f2 = *(const v4f*)(p + 16);
            v4f f3 = *(const v4f*)(p + 20);
#pragma unroll
            for (int j = 0; j < 4; ++j) {
                a[j]      = (_Float16)f0[j];
                a[4 + j]  = (_Float16)f1[j];
                a[8 + j]  = (_Float16)f2[j];
                a[12 + j] = (_Float16)f3[j];
            }
        }
#pragma unroll
        for (int ct = 0; ct < 8; ++ct) {
            const _Float16* bp = Bs + ((ct * KSTEPS + ks) * 32 + lane) * 16;
            V16HU ub;
            ub.f[0] = *(const v4f*)(bp);
            ub.f[1] = *(const v4f*)(bp + 8);
            acc[ct] = __builtin_amdgcn_wmma_f32_16x16x32_f16(
                false, a, false, ub.h, (short)0, acc[ct], false, false);
        }
    }

    const int orow = rt * 16 + hi16 * 8;           // D rows this lane holds: orow+v
#pragma unroll
    for (int ct = 0; ct < 8; ++ct) {
        const int col = ct * 16 + ln15;
#pragma unroll
        for (int v = 0; v < 8; ++v) {
            float x = acc[ct][v];
            if (relu) x = fmaxf(x, 0.f);
            size_t idx = (size_t)(orow + v) * 128 + col;
            if (outF) outF[idx] = x;
            if (outH) outH[idx] = (_Float16)x;
        }
    }
}

// ---------------------------------------------------------------------------
// Edge scatter-add: m[dst] += h[src]; one lane handles 4 channels (float4).
// ---------------------------------------------------------------------------
__global__ void k_scatter(const float* __restrict__ h, const int* __restrict__ src,
                          const int* __restrict__ dst, float* __restrict__ m, int E) {
    int tid = blockIdx.x * blockDim.x + threadIdx.x;
    if (tid >= E * 32) return;
    int e  = tid >> 5;
    int c0 = (tid & 31) * 4;
    int s = src[e], d = dst[e];
    v4f v = *(const v4f*)(h + (size_t)s * 128 + c0);
    float* p = m + (size_t)d * 128 + c0;
    atomicAdd(p + 0, v.x);
    atomicAdd(p + 1, v.y);
    atomicAdd(p + 2, v.z);
    atomicAdd(p + 3, v.w);
}

// ---------------------------------------------------------------------------
// BatchNorm statistics / finalize / apply+relu
// ---------------------------------------------------------------------------
__global__ void k_bnstats(const float* __restrict__ z, float* __restrict__ sums, int N) {
    int c = threadIdx.x;  // 128
    int rows_per = (N + gridDim.x - 1) / gridDim.x;
    int r0 = blockIdx.x * rows_per;
    int r1 = min(N, r0 + rows_per);
    float s = 0.f, q = 0.f;
    for (int r = r0; r < r1; ++r) {
        float v = z[(size_t)r * 128 + c];
        s += v; q += v * v;
    }
    atomicAdd(&sums[c], s);
    atomicAdd(&sums[128 + c], q);
}

__global__ void k_bnfin(const float* __restrict__ sums,
                        const float* __restrict__ gamma, const float* __restrict__ beta,
                        float* __restrict__ ss, int N) {
    int c = threadIdx.x;  // 128
    float inv = 1.f / (float)N;
    float mean = sums[c] * inv;
    float var  = sums[128 + c] * inv - mean * mean;
    float sc   = gamma[c] * rsqrtf(var + BN_EPS);
    ss[c]       = sc;
    ss[128 + c] = beta[c] - mean * sc;
}

__global__ void k_bnrelu(const float* __restrict__ z, const float* __restrict__ ss,
                         float* __restrict__ h, int total) {
    int i = blockIdx.x * blockDim.x + threadIdx.x;
    if (i >= total) return;
    int c = i & 127;
    h[i] = fmaxf(z[i] * ss[c] + ss[128 + c], 0.f);
}

// ---------------------------------------------------------------------------
// Graph pooling: pooled[batch[n]] += h[n]
// ---------------------------------------------------------------------------
__global__ void k_pool(const float* __restrict__ h, const int* __restrict__ batch,
                       float* __restrict__ pooled, int N) {
    int tid = blockIdx.x * blockDim.x + threadIdx.x;
    if (tid >= N * 32) return;
    int n  = tid >> 5;
    int c0 = (tid & 31) * 4;
    v4f v = *(const v4f*)(h + (size_t)n * 128 + c0);
    float* p = pooled + (size_t)batch[n] * 128 + c0;
    atomicAdd(p + 0, v.x);
    atomicAdd(p + 1, v.y);
    atomicAdd(p + 2, v.z);
    atomicAdd(p + 3, v.w);
}

// ---------------------------------------------------------------------------
// Readout tail: out[g] = dot(r[g], roW2) + rob2 (one wave per graph)
// ---------------------------------------------------------------------------
__global__ void k_out(const float* __restrict__ r, const float* __restrict__ w2,
                      const float* __restrict__ b2, float* __restrict__ out) {
    int g = blockIdx.x;
    int l = threadIdx.x;  // 32
    float s = 0.f;
#pragma unroll
    for (int j = 0; j < 4; ++j) {
        int c = l * 4 + j;
        s += r[(size_t)g * 128 + c] * w2[c];
    }
#pragma unroll
    for (int off = 16; off > 0; off >>= 1) s += __shfl_down(s, off, 32);
    if (l == 0) out[g] = s + b2[0];
}

// ---------------------------------------------------------------------------
// Host launcher
// ---------------------------------------------------------------------------
extern "C" void kernel_launch(void* const* d_in, const int* in_sizes, int n_in,
                              void* d_out, int out_size, void* d_ws, size_t ws_size,
                              hipStream_t stream) {
    const int N = in_sizes[0];
    const int E = in_sizes[2] / 2;
    const int G = out_size;
    const int Lnum = in_sizes[14] / 128;  // convb1 = [L,128]

    const int*   x        = (const int*)  d_in[0];
    const float* pe       = (const float*)d_in[1];
    const int*   ei       = (const int*)  d_in[2];
    const int*   batch    = (const int*)  d_in[3];
    const float* atom_emb = (const float*)d_in[4];
    const float* peW1     = (const float*)d_in[5];
    const float* peb1     = (const float*)d_in[6];
    const float* peW2     = (const float*)d_in[7];
    const float* peb2     = (const float*)d_in[8];
    const float* pe_gamma = (const float*)d_in[9];
    const float* pe_beta  = (const float*)d_in[10];
    const float* inW      = (const float*)d_in[11];
    const float* inb      = (const float*)d_in[12];
    const float* convW1   = (const float*)d_in[13];
    const float* convb1   = (const float*)d_in[14];
    const float* convW2   = (const float*)d_in[15];
    const float* convb2   = (const float*)d_in[16];
    const float* bn_gamma = (const float*)d_in[17];
    const float* bn_beta  = (const float*)d_in[18];
    const float* roW1     = (const float*)d_in[19];
    const float* rob1     = (const float*)d_in[20];
    const float* roW2     = (const float*)d_in[21];
    const float* rob2     = (const float*)d_in[22];
    float* out = (float*)d_out;

    const int* src = ei;
    const int* dst = ei + E;

    // Workspace carve-up (all offsets 256B aligned)
    char* W = (char*)d_ws;
    size_t off = 0;
    auto take = [&](size_t bytes) -> void* {
        void* p = W + off;
        off += (bytes + 255) & ~(size_t)255;
        return p;
    };
    float*    h       = (float*)   take((size_t)N * 128 * 4);
    float*    mz      = (float*)   take((size_t)N * 128 * 4);  // h_in / m / z (time-shared)
    _Float16* t16     = (_Float16*)take((size_t)N * 128 * 2);
    float*    hp      = (float*)   take((size_t)N * 16 * 4);
    float*    pooled  = (float*)   take((size_t)G * 128 * 4);
    float*    r32     = (float*)   take((size_t)G * 128 * 4);
    float*    pe_sums = (float*)   take(32 * 4);
    float*    pe_ss   = (float*)   take(32 * 4);
    float*    bn_sums = (float*)   take(256 * 4);
    float*    bn_ss   = (float*)   take(256 * 4);
    _Float16* inWp    = (_Float16*)take(8 * 3 * 512 * 2);            // K=96 packed
    _Float16* c1p     = (_Float16*)take((size_t)Lnum * 16384 * 2);
    _Float16* c2p     = (_Float16*)take((size_t)Lnum * 16384 * 2);
    _Float16* roWp    = (_Float16*)take(16384 * 2);
    (void)ws_size; (void)n_in;

    const int nRT  = N / 16;                 // 2500 row tiles
    const int gRT  = G / 16;                 // 128 row tiles
    const int gemmBlocks  = (nRT + 3) / 4;   // 4 waves (row tiles) per block
    const int gemmBlocksG = (gRT + 3) / 4;

    // --- PE MLP + its BatchNorm ---
    k_zero<<<1, 64, 0, stream>>>(pe_sums, 32);
    k_pe<<<(N + 255) / 256, 256, 0, stream>>>(pe, peW1, peb1, peW2, peb2, hp, pe_sums, N);
    k_pe_fin<<<1, 16, 0, stream>>>(pe_sums, pe_gamma, pe_beta, pe_ss, N);

    // --- Build concatenated input h_in[N,96] (zero-padded to K=96) ---
    k_embed<<<N, 96, 0, stream>>>(x, atom_emb, hp, pe_ss, mz);

    // --- Pack all weight matrices into WMMA B-operand order ---
    k_pack<<<(8 * 3 * 512 + 255) / 256, 256, 0, stream>>>(inW, inWp, 80, 3);
    for (int i = 0; i < Lnum; ++i) {
        k_pack<<<64, 256, 0, stream>>>(convW1 + (size_t)i * 16384, c1p + (size_t)i * 16384, 128, 4);
        k_pack<<<64, 256, 0, stream>>>(convW2 + (size_t)i * 16384, c2p + (size_t)i * 16384, 128, 4);
    }
    k_pack<<<64, 256, 0, stream>>>(roW1, roWp, 128, 4);

    // --- Input projection: h = h_in @ inW + inb  (no relu) ---
    gemm_wmma<0, 3><<<gemmBlocks, 128, 0, stream>>>(mz, inWp, inb, h, nullptr, nRT, 0);

    // --- GIN layers ---
    for (int i = 0; i < Lnum; ++i) {
        // m = h (self term), then scatter-add neighbors
        hipMemcpyAsync(mz, h, (size_t)N * 128 * 4, hipMemcpyDeviceToDevice, stream);
        k_scatter<<<((size_t)E * 32 + 255) / 256, 256, 0, stream>>>(h, src, dst, mz, E);
        // t = relu(m @ W1 + b1) -> f16
        gemm_wmma<0, 4><<<gemmBlocks, 128, 0, stream>>>(
            mz, c1p + (size_t)i * 16384, convb1 + i * 128, nullptr, t16, nRT, 1);
        // z = t @ W2 + b2 -> f32 (reuses mz)
        gemm_wmma<1, 4><<<gemmBlocks, 128, 0, stream>>>(
            t16, c2p + (size_t)i * 16384, convb2 + i * 128, mz, nullptr, nRT, 0);
        // h = relu(bn(z))
        k_zero<<<1, 256, 0, stream>>>(bn_sums, 256);
        k_bnstats<<<256, 128, 0, stream>>>(mz, bn_sums, N);
        k_bnfin<<<1, 128, 0, stream>>>(bn_sums, bn_gamma + i * 128, bn_beta + i * 128, bn_ss, N);
        k_bnrelu<<<((size_t)N * 128 + 255) / 256, 256, 0, stream>>>(mz, bn_ss, h, N * 128);
    }

    // --- Global add-pool + readout MLP ---
    k_zero<<<(G * 128 + 255) / 256, 256, 0, stream>>>(pooled, G * 128);
    k_pool<<<((size_t)N * 32 + 255) / 256, 256, 0, stream>>>(h, batch, pooled, N);
    gemm_wmma<0, 4><<<gemmBlocksG, 128, 0, stream>>>(pooled, roWp, rob1, r32, nullptr, gRT, 1);
    k_out<<<G, 32, 0, stream>>>(r32, roW2, rob2, out);
}
